// FrameTransformerDecoder_66202625900955
// MI455X (gfx1250) — compile-verified, matrix operand order checked by reading, not run on hardware
//
// FrameTransformerDecoder for MI455X (gfx1250, wave32, WMMA).
// All heavy einsums -> one generic bf16-WMMA GEMM (f32 accumulate), fp32
// operands converted to bf16 (add-0x8000 + v_perm_b32 pack) while staging
// into LDS. Double-buffered LDS software pipeline: global loads for tile k+1
// issue before the WMMAs of tile k, conversion+LDS stores land after, one
// barrier per K-step. Tile: 128x128/WG, K-step 32, 8 waves, 2x4 WMMA/wave.
#include <hip/hip_runtime.h>
#include <hip/hip_bf16.h>

typedef __attribute__((ext_vector_type(16))) __bf16 v16bf;
typedef __attribute__((ext_vector_type(2)))  __bf16 v2bf;
typedef __attribute__((ext_vector_type(8)))  float  v8f;

#define BB 2
#define CC 4
#define FF 1024
#define WW 512
#define NHH 8
#define HDD 128

// pack two fp32 into bf16x2: round-to-nearest (ties away) via +0x8000, then
// one v_perm_b32 selects the two high halves. 3 VALU ops total.
static __device__ __forceinline__ unsigned int pack2bf(float a, float b) {
#if __has_builtin(__builtin_amdgcn_cvt_pk_bf16_f32)
  v2bf r = __builtin_amdgcn_cvt_pk_bf16_f32(a, b);
  return __builtin_bit_cast(unsigned int, r);
#else
  unsigned int ua = __builtin_bit_cast(unsigned int, a) + 0x8000u;
  unsigned int ub = __builtin_bit_cast(unsigned int, b) + 0x8000u;
  // dst = { ub[31:24], ub[23:16], ua[31:24], ua[23:16] }
  return __builtin_amdgcn_perm(ub, ua, 0x07060302u);
#endif
}

union FragAB { uint4 u[2]; v16bf v; };

// ---------------------------------------------------------------------------
// Generic slice-batched GEMM: C[z] = alpha * A[za] * B[z] (+ epilogue)
//   A: fp32, M x K, row-major, lda. za = aMod ? z % aMod : z.
//   B: fp32. bTrans==0: K x N row-major. bTrans==1: N x K row-major (B^T).
//   EPI: 0 = store, 1 = relu(x)^2, 2 = add residual R (same layout as C).
// M % 128 == 0, N % 128 == 0, K % 32 == 0 by construction of the model.
// LDS bf16 tile layout: [row][16 dwords], dword j = K-pair (2j, 2j+1).
// ---------------------------------------------------------------------------
template<int EPI>
__global__ __launch_bounds__(256)
void gemm_bf16(const float* __restrict__ A, long aStride, int aMod,
               const float* __restrict__ B, long bStride,
               float* __restrict__ C, long cStride,
               const float* __restrict__ R, long rStride,
               int M, int N, int K,
               int lda, int ldb, int ldc,
               int bTrans, float alpha)
{
  __shared__ unsigned int sA[2][128 * 16];
  __shared__ unsigned int sB[2][128 * 16];

  const int z = blockIdx.z;
  const float* Ab = A + (long)(aMod ? (z % aMod) : z) * aStride;
  const float* Bb = B + (long)z * bStride;
  float*       Cb = C + (long)z * cStride;
  const float* Rb = (EPI == 2) ? (R + (long)z * rStride) : nullptr;

  const int tid      = threadIdx.x;
  const int lane     = tid & 31;
  const int wave     = tid >> 5;
  const int laneHalf = lane >> 4;     // 0: K{0..7,16..23}, 1: K{8..15,24..31}
  const int laneRow  = lane & 15;
  const int waveM = (wave & 3) * 32;  // 4 waves along M
  const int waveN = (wave >> 2) * 64; // 2 waves along N
  const int m0g = blockIdx.x * 128;
  const int n0g = blockIdx.y * 128;

  const v8f zero8 = {0.f,0.f,0.f,0.f,0.f,0.f,0.f,0.f};
  v8f acc[2][4];
  #pragma unroll
  for (int i = 0; i < 2; ++i)
    #pragma unroll
    for (int j = 0; j < 4; ++j) acc[i][j] = zero8;

  float4 ra[4], rb[4];

  auto loadA = [&](int k0) {
    #pragma unroll
    for (int i = 0; i < 4; ++i) {
      int idx = tid + i * 256;
      int r = idx >> 3, c4 = idx & 7;
      ra[i] = *(const float4*)(Ab + (long)(m0g + r) * lda + k0 + c4 * 4);
    }
  };
  auto storeA = [&](unsigned int* s) {
    #pragma unroll
    for (int i = 0; i < 4; ++i) {
      int idx = tid + i * 256;
      int r = idx >> 3, c4 = idx & 7;
      s[r * 16 + c4 * 2]     = pack2bf(ra[i].x, ra[i].y);
      s[r * 16 + c4 * 2 + 1] = pack2bf(ra[i].z, ra[i].w);
    }
  };
  auto loadB = [&](int k0) {
    if (bTrans) {
      #pragma unroll
      for (int i = 0; i < 4; ++i) {
        int idx = tid + i * 256;
        int r = idx >> 3, c4 = idx & 7;
        rb[i] = *(const float4*)(Bb + (long)(n0g + r) * ldb + k0 + c4 * 4);
      }
    } else {
      // tile 32(k) x 128(n): 16 k-pairs x 32 n-quads, 2 tasks per thread
      #pragma unroll
      for (int i = 0; i < 2; ++i) {
        int idx = tid + i * 256;
        int kp = idx >> 5, n4 = idx & 31;
        rb[2 * i]     = *(const float4*)(Bb + (long)(k0 + 2 * kp)     * ldb + n0g + n4 * 4);
        rb[2 * i + 1] = *(const float4*)(Bb + (long)(k0 + 2 * kp + 1) * ldb + n0g + n4 * 4);
      }
    }
  };
  auto storeB = [&](unsigned int* s) {
    if (bTrans) {
      #pragma unroll
      for (int i = 0; i < 4; ++i) {
        int idx = tid + i * 256;
        int r = idx >> 3, c4 = idx & 7;
        s[r * 16 + c4 * 2]     = pack2bf(rb[i].x, rb[i].y);
        s[r * 16 + c4 * 2 + 1] = pack2bf(rb[i].z, rb[i].w);
      }
    } else {
      #pragma unroll
      for (int i = 0; i < 2; ++i) {
        int idx = tid + i * 256;
        int kp = idx >> 5, n4 = idx & 31;
        const float4 v0 = rb[2 * i], v1 = rb[2 * i + 1];
        s[(n4 * 4 + 0) * 16 + kp] = pack2bf(v0.x, v1.x);
        s[(n4 * 4 + 1) * 16 + kp] = pack2bf(v0.y, v1.y);
        s[(n4 * 4 + 2) * 16 + kp] = pack2bf(v0.z, v1.z);
        s[(n4 * 4 + 3) * 16 + kp] = pack2bf(v0.w, v1.w);
      }
    }
  };

  // prologue: stage tile 0
  loadA(0); loadB(0);
  storeA(sA[0]); storeB(sB[0]);
  __syncthreads();

  int cur = 0;
  for (int k0 = 0; k0 < K; k0 += 32) {
    const bool more = (k0 + 32 < K);
    // issue next tile's global loads first (latency hides under WMMAs)
    if (more) {
      loadA(k0 + 32);
      loadB(k0 + 32);
      // L2 prefetch one tile further ahead (global_prefetch_b8)
      if (k0 + 64 < K && tid < 128)
        __builtin_prefetch(Ab + (long)(m0g + tid) * lda + k0 + 64, 0, 1);
    }

    // ---- fragments + WMMA from current LDS buffer ----
    const unsigned int* cA = sA[cur];
    const unsigned int* cB = sB[cur];
    v16bf afr[2], bfr[4];
    #pragma unroll
    for (int mt = 0; mt < 2; ++mt) {
      int row = waveM + mt * 16 + laneRow;
      FragAB f;
      f.u[0] = *(const uint4*)&cA[row * 16 + laneHalf * 4];
      f.u[1] = *(const uint4*)&cA[row * 16 + laneHalf * 4 + 8];
      afr[mt] = f.v;
    }
    #pragma unroll
    for (int nt = 0; nt < 4; ++nt) {
      int row = waveN + nt * 16 + laneRow;
      FragAB f;
      f.u[0] = *(const uint4*)&cB[row * 16 + laneHalf * 4];
      f.u[1] = *(const uint4*)&cB[row * 16 + laneHalf * 4 + 8];
      bfr[nt] = f.v;
    }
    #pragma unroll
    for (int mt = 0; mt < 2; ++mt)
      #pragma unroll
      for (int nt = 0; nt < 4; ++nt)
        acc[mt][nt] = __builtin_amdgcn_wmma_f32_16x16x32_bf16(
            false, afr[mt], false, bfr[nt], (short)0, acc[mt][nt], false, false);

    // ---- convert + store next tile into the other LDS buffer ----
    if (more) {
      storeA(sA[cur ^ 1]);
      storeB(sB[cur ^ 1]);
    }
    __syncthreads();
    cur ^= 1;
  }

  // ---- epilogue: C/D layout: vgpr vr -> M = vr + 8*laneHalf, N = laneRow ----
  #pragma unroll
  for (int mt = 0; mt < 2; ++mt) {
    int rbase = m0g + waveM + mt * 16 + laneHalf * 8;
    #pragma unroll
    for (int nt = 0; nt < 4; ++nt) {
      int col = n0g + waveN + nt * 16 + laneRow;
      #pragma unroll
      for (int vr = 0; vr < 8; ++vr) {
        float v = acc[mt][nt][vr] * alpha;
        if (EPI == 1) { v = fmaxf(v, 0.f); v = v * v; }
        if (EPI == 2) { v += Rb[(long)(rbase + vr) * ldc + col]; }
        Cb[(long)(rbase + vr) * ldc + col] = v;
      }
    }
  }
}

// ---------------------------------------------------------------------------
// conv1x1 channel embed: y[b,c,f,w] = eb[c] + sum_ci ew[c,ci] * x[b,ci,f,w]
// ---------------------------------------------------------------------------
__global__ __launch_bounds__(256)
void embed_kernel(const float* __restrict__ x, const float* __restrict__ ew,
                  const float* __restrict__ eb, float* __restrict__ y)
{
  const long FWl = (long)FF * WW;
  long i = (long)blockIdx.x * 256 + threadIdx.x;      // over B*F*W
  if (i >= (long)BB * FWl) return;
  int b = (int)(i / FWl); long fw = i % FWl;
  float in[CC];
  #pragma unroll
  for (int ci = 0; ci < CC; ++ci) in[ci] = x[((long)b * CC + ci) * FWl + fw];
  #pragma unroll
  for (int c = 0; c < CC; ++c) {
    float acc = eb[c];
    #pragma unroll
    for (int ci = 0; ci < CC; ++ci) acc += ew[c * CC + ci] * in[ci];
    y[((long)b * CC + c) * FWl + fw] = acc;
  }
}

// ---------------------------------------------------------------------------
// frame_norm over f (axis=2), per (b,c,w) column; consecutive threads take
// consecutive w -> coalesced strided sweeps.
// ---------------------------------------------------------------------------
__global__ __launch_bounds__(256)
void frame_norm_kernel(const float* __restrict__ x, const float* __restrict__ g,
                       const float* __restrict__ bta, float* __restrict__ y)
{
  int idx = blockIdx.x * 256 + threadIdx.x;           // over B*C*W = 4096
  if (idx >= BB * CC * WW) return;
  int w = idx % WW; int bc = idx / WW; int c = bc % CC;
  const float* col = x + (long)bc * FF * WW + w;
  float s = 0.f, s2 = 0.f;
  for (int f = 0; f < FF; ++f) { float v = col[(long)f * WW]; s += v; s2 += v * v; }
  float m = s * (1.f / FF);
  float var = s2 * (1.f / FF) - m * m;
  float inv = rsqrtf(var + 1e-5f);
  float* out = y + (long)bc * FF * WW + w;
  for (int f = 0; f < FF; ++f) {
    float v = (col[(long)f * WW] - m) * inv;
    out[(long)f * WW] = v * g[c * FF + f] + bta[c * FF + f];
  }
}

// ---------------------------------------------------------------------------
// conv1x3 over W with channel mix (C=4), zero pad: z = cb + sum_{ci,t} cw*y
// ---------------------------------------------------------------------------
__global__ __launch_bounds__(256)
void conv1x3_kernel(const float* __restrict__ y, const float* __restrict__ cw,
                    const float* __restrict__ cb, float* __restrict__ z)
{
  long i = (long)blockIdx.x * 256 + threadIdx.x;      // over B*C*F*W
  const long tot = (long)BB * CC * FF * WW;
  if (i >= tot) return;
  int w = (int)(i % WW); long t = i / WW;
  int f = (int)(t % FF); t /= FF;
  int c = (int)(t % CC); int b = (int)(t / CC);
  float acc = cb[c];
  #pragma unroll
  for (int ci = 0; ci < CC; ++ci) {
    const float* row = y + (((long)(b * CC + ci)) * FF + f) * WW;
    float l = (w > 0)      ? row[w - 1] : 0.f;
    float m = row[w];
    float r = (w < WW - 1) ? row[w + 1] : 0.f;
    const float* wt = cw + (c * CC + ci) * 3;
    acc += wt[0] * l + wt[1] * m + wt[2] * r;
  }
  z[i] = acc;
}

// ---------------------------------------------------------------------------
// to_heads (+ optional rotary): (b,c,f=n*128+d,w) -> (slice=(b*C+c)*8+n, w, d)
// rotary: even/odd pairs, angle = w * 10000^(-2i/128)
// ---------------------------------------------------------------------------
__global__ __launch_bounds__(256)
void heads_kernel(const float* __restrict__ z, float* __restrict__ q, int doRot)
{
  long i = (long)blockIdx.x * 256 + threadIdx.x;      // over slices*W*(HD/2)
  const long tot = (long)BB * CC * NHH * WW * (HDD / 2);
  if (i >= tot) return;
  int dp = (int)(i % (HDD / 2)); long t = i / (HDD / 2);
  int w = (int)(t % WW); int s = (int)(t / WW);
  int n = s % NHH; int bc = s / NHH;
  const float* src = z + ((long)bc * FF + n * HDD + dp * 2) * WW + w;
  float xe = src[0], xo = src[WW];
  if (doRot) {
    float inv = expf(-((float)(2 * dp) / (float)HDD) * logf(10000.f));
    float ang = (float)w * inv;
    float cs = cosf(ang), sn = sinf(ang);
    float oe = xe * cs - xo * sn;
    float oo = xo * cs + xe * sn;
    xe = oe; xo = oo;
  }
  float* dst = q + ((long)s * WW + w) * HDD + dp * 2;
  dst[0] = xe; dst[1] = xo;
}

// (slice, w, d) -> (b,c,f=n*128+d,w)
__global__ __launch_bounds__(256)
void heads_inv_kernel(const float* __restrict__ a, float* __restrict__ out)
{
  long i = (long)blockIdx.x * 256 + threadIdx.x;      // over slices*W*HD
  const long tot = (long)BB * CC * NHH * WW * HDD;
  if (i >= tot) return;
  int d = (int)(i % HDD); long t = i / HDD;
  int w = (int)(t % WW); int s = (int)(t / WW);
  int n = s % NHH; int bc = s / NHH;
  out[((long)bc * FF + n * HDD + d) * WW + w] = a[i];
}

// ---------------------------------------------------------------------------
// row softmax, row length 512, one workgroup (8 waves) per row
// ---------------------------------------------------------------------------
__global__ __launch_bounds__(256)
void softmax_kernel(float* __restrict__ S)
{
  float* row = S + (long)blockIdx.x * WW;
  int tid = threadIdx.x;
  float a = row[tid], b = row[tid + 256];
  __shared__ float red[8];

  float mx = fmaxf(a, b);
  #pragma unroll
  for (int o = 16; o > 0; o >>= 1) mx = fmaxf(mx, __shfl_xor(mx, o, 32));
  if ((tid & 31) == 0) red[tid >> 5] = mx;
  __syncthreads();
  mx = red[0];
  #pragma unroll
  for (int k = 1; k < 8; ++k) mx = fmaxf(mx, red[k]);
  __syncthreads();

  float e0 = __expf(a - mx), e1 = __expf(b - mx);
  float sm = e0 + e1;
  #pragma unroll
  for (int o = 16; o > 0; o >>= 1) sm += __shfl_xor(sm, o, 32);
  if ((tid & 31) == 0) red[tid >> 5] = sm;
  __syncthreads();
  sm = red[0];
  #pragma unroll
  for (int k = 1; k < 8; ++k) sm += red[k];
  float rinv = 1.f / sm;
  row[tid] = e0 * rinv;
  row[tid + 256] = e1 * rinv;
}

// ---------------------------------------------------------------------------
static inline void launch_gemm(hipStream_t st, int epi,
    const float* A, long aS, int aMod, const float* B, long bS,
    float* C, long cS, const float* R, long rS,
    int M, int N, int K, int lda, int ldb, int ldc,
    int bTrans, float alpha, int slices)
{
  dim3 g(M / 128, N / 128, slices), b(256);
  switch (epi) {
    case 0: gemm_bf16<0><<<g, b, 0, st>>>(A, aS, aMod, B, bS, C, cS, R, rS, M, N, K, lda, ldb, ldc, bTrans, alpha); break;
    case 1: gemm_bf16<1><<<g, b, 0, st>>>(A, aS, aMod, B, bS, C, cS, R, rS, M, N, K, lda, ldb, ldc, bTrans, alpha); break;
    default: gemm_bf16<2><<<g, b, 0, st>>>(A, aS, aMod, B, bS, C, cS, R, rS, M, N, K, lda, ldb, ldc, bTrans, alpha); break;
  }
}

extern "C" void kernel_launch(void* const* d_in, const int* in_sizes, int n_in,
                              void* d_out, int out_size, void* d_ws, size_t ws_size,
                              hipStream_t stream)
{
  (void)in_sizes; (void)n_in; (void)out_size; (void)ws_size;
  const float* x    = (const float*)d_in[0];
  const float* skip = (const float*)d_in[1];
  const float* esw  = (const float*)d_in[2];
  const float* esb  = (const float*)d_in[3];
  const float* ekw  = (const float*)d_in[4];
  const float* ekb  = (const float*)d_in[5];
  const float* n1g  = (const float*)d_in[6];
  const float* n1b  = (const float*)d_in[7];
  const float* n2g  = (const float*)d_in[8];
  const float* n2b  = (const float*)d_in[9];
  const float* n3g  = (const float*)d_in[10];
  const float* n3b  = (const float*)d_in[11];
  const float* l1w  = (const float*)d_in[32];
  const float* l2w  = (const float*)d_in[33];

  float* ws = (float*)d_ws;
  const long MSL = (long)BB * CC * FF * WW;   // 4M floats per activation buffer
  float* xA  = ws;                            // current x
  float* skE = ws + 1 * MSL;                  // embedded skip
  float* t0  = ws + 2 * MSL;                  // normed x / attn f,w
  float* t1  = ws + 3 * MSL;                  // q proj / q heads
  float* t2  = ws + 4 * MSL;                  // k proj / k heads
  float* t3  = ws + 5 * MSL;                  // v proj / v heads
  float* t4  = ws + 6 * MSL;                  // conv scratch / attn heads out
  float* Ssc = ws + 7 * MSL;                  // 16M floats: scores / MLP hidden

  const long SL = (long)FF * WW;              // 524288, per-(b,c) matrix
  const long HS = (long)WW * HDD;             // 65536, per-head matrix
  const long SS = (long)WW * WW;              // 262144, per-head score matrix
  const long totBFW  = (long)BB * FF * WW;
  const long totBCFW = (long)BB * CC * FF * WW;
  const long totHp   = (long)BB * CC * NHH * WW * (HDD / 2);
  const long totH    = (long)BB * CC * NHH * WW * HDD;

  // channel embeds
  embed_kernel<<<(int)((totBFW + 255) / 256), 256, 0, stream>>>(x, esw, esb, xA);
  embed_kernel<<<(int)((totBFW + 255) / 256), 256, 0, stream>>>(skip, ekw, ekb, skE);

  auto run_attn = [&](int base, const float* ng, const float* nb, const float* kvsrc) {
    const float* qw  = (const float*)d_in[base + 0];
    const float* kw  = (const float*)d_in[base + 1];
    const float* vw  = (const float*)d_in[base + 2];
    const float* qcw = (const float*)d_in[base + 3];
    const float* kcw = (const float*)d_in[base + 4];
    const float* vcw = (const float*)d_in[base + 5];
    const float* qcb = (const float*)d_in[base + 6];
    const float* kcb = (const float*)d_in[base + 7];
    const float* vcb = (const float*)d_in[base + 8];
    const float* ow  = (const float*)d_in[base + 9];

    frame_norm_kernel<<<16, 256, 0, stream>>>(xA, ng, nb, t0);
    // projections: out[f,w] = W[c] @ xn  (8 slices, weights shared over b)
    launch_gemm(stream, 0, qw, (long)FF * FF, CC, t0,    SL, t1, SL, nullptr, 0,
                FF, WW, FF, FF, WW, WW, 0, 1.f, BB * CC);
    launch_gemm(stream, 0, kw, (long)FF * FF, CC, kvsrc, SL, t2, SL, nullptr, 0,
                FF, WW, FF, FF, WW, WW, 0, 1.f, BB * CC);
    launch_gemm(stream, 0, vw, (long)FF * FF, CC, kvsrc, SL, t3, SL, nullptr, 0,
                FF, WW, FF, FF, WW, WW, 0, 1.f, BB * CC);
    // conv1x3 + bias, then head split (+rotary for q,k)
    conv1x3_kernel<<<(int)((totBCFW + 255) / 256), 256, 0, stream>>>(t1, qcw, qcb, t4);
    heads_kernel<<<(int)((totHp + 255) / 256), 256, 0, stream>>>(t4, t1, 1);
    conv1x3_kernel<<<(int)((totBCFW + 255) / 256), 256, 0, stream>>>(t2, kcw, kcb, t4);
    heads_kernel<<<(int)((totHp + 255) / 256), 256, 0, stream>>>(t4, t2, 1);
    conv1x3_kernel<<<(int)((totBCFW + 255) / 256), 256, 0, stream>>>(t3, vcw, vcb, t4);
    heads_kernel<<<(int)((totHp + 255) / 256), 256, 0, stream>>>(t4, t3, 0);
    // S = (q @ k^T) / sqrt(1024)   (64 head slices; k supplied as B^T)
    launch_gemm(stream, 0, t1, HS, 0, t2, HS, Ssc, SS, nullptr, 0,
                WW, WW, HDD, HDD, HDD, WW, 1, 0.03125f, BB * CC * NHH);
    softmax_kernel<<<BB * CC * NHH * WW, 256, 0, stream>>>(Ssc);
    // A = P @ v
    launch_gemm(stream, 0, Ssc, SS, 0, t3, HS, t4, HS, nullptr, 0,
                WW, HDD, WW, WW, HDD, HDD, 0, 1.f, BB * CC * NHH);
    heads_inv_kernel<<<(int)((totH + 255) / 256), 256, 0, stream>>>(t4, t0);
    // out projection with fused residual: xA = xA + ow @ attn
    launch_gemm(stream, 2, ow, (long)FF * FF, CC, t0, SL, xA, SL, xA, SL,
                FF, WW, FF, FF, WW, WW, 0, 1.f, BB * CC);
  };

  run_attn(12, n1g, n1b, t0);    // self-attn: k/v from normed x
  run_attn(22, n2g, n2b, skE);   // skip-attn: k/v from embedded skip

  // MLP: h = relu(l1 @ xn)^2 ; out = xA + l2 @ h   (written straight to d_out)
  frame_norm_kernel<<<16, 256, 0, stream>>>(xA, n3g, n3b, t0);
  launch_gemm(stream, 1, l1w, (long)4 * FF * FF, CC, t0, SL,
              Ssc, (long)4 * FF * WW, nullptr, 0,
              4 * FF, WW, FF, FF, WW, WW, 0, 1.f, BB * CC);
  launch_gemm(stream, 2, l2w, (long)4 * FF * FF, CC, Ssc, (long)4 * FF * WW,
              (float*)d_out, SL, xA, SL,
              FF, WW, 4 * FF, 4 * FF, WW, WW, 0, 1.f, BB * CC);
}
